// MMDLoss_9268539424987
// MI455X (gfx1250) — matrix operand order repaired and checked by reading, not compiled, not run on hardware
//
#include <hip/hip_runtime.h>

typedef __attribute__((ext_vector_type(16))) _Float16 v16h;
typedef __attribute__((ext_vector_type(8)))  _Float16 v8h;
typedef __attribute__((ext_vector_type(8)))  float    v8f;

#define NTOT   16384      // total rows (x + y)
#define NHALF  8192       // rows of x (== rows of y)
#define DIM    64
#define TTILE  1024       // NTOT / 16
#define TJB    8          // wave-tiles per block along j
#define GRIDY  (TTILE / TJB)        // 128
#define NPART  (TTILE * GRIDY)      // 131072 block partials

// workspace layout (float offsets)
#define SQ_F     0                       // sq[i] row norms       (16384)
#define COLP_F   16384                   // per-block column sums (64*64)
#define S1P_F    20480                   // per-block sq sums     (64)
#define SCAL_F   20544                   // coef = 1/(4*bw*ln2)
#define PART_F   20608                   // main-pass block partials (131072)
#define PART2_F  151680                  // stage-2 partials (256)
#define ZH_F     151936                  // f16 hi split of z (1M halves = 512K floats)
#define ZL_F     676224                  // f16 lo split of z

// Raw v_exp_f32: our exponent is in [-~20, 0], so OCML's denorm/range fixup is dead weight.
#if defined(__has_builtin)
#  if __has_builtin(__builtin_amdgcn_exp2f)
#    define FAST_EXP2(x) __builtin_amdgcn_exp2f(x)
#  endif
#endif
#ifndef FAST_EXP2
#  define FAST_EXP2(x) exp2f(x)
#endif

__device__ __forceinline__ const float* zrow(const float* x, const float* y, int r) {
  return (r < NHALF) ? (x + (size_t)r * DIM) : (y + (size_t)(r - NHALF) * DIM);
}

// Pass 1a: per-row squared norms, f16 hi/lo split of z, per-block sum of norms.
__global__ void k_rows(const float* __restrict__ x, const float* __restrict__ y,
                       float* __restrict__ wsf) {
  const int i = blockIdx.x * 256 + threadIdx.x;
  const float4* z4 = (const float4*)zrow(x, y, i);
  _Float16* zh = (_Float16*)(wsf + ZH_F) + (size_t)i * DIM;
  _Float16* zl = (_Float16*)(wsf + ZL_F) + (size_t)i * DIM;
  float sq = 0.f;
  #pragma unroll
  for (int c = 0; c < 8; ++c) {
    float4 f0 = z4[2 * c], f1 = z4[2 * c + 1];
    float f[8] = {f0.x, f0.y, f0.z, f0.w, f1.x, f1.y, f1.z, f1.w};
    v8h h, l;
    #pragma unroll
    for (int e = 0; e < 8; ++e) {
      sq += f[e] * f[e];
      _Float16 hv = (_Float16)f[e];
      h[e] = hv;
      l[e] = (_Float16)(f[e] - (float)hv);
    }
    ((v8h*)zh)[c] = h;
    ((v8h*)zl)[c] = l;
  }
  wsf[SQ_F + i] = sq;
  __shared__ float sm[256];
  sm[threadIdx.x] = sq;
  __syncthreads();
  for (int s = 128; s > 0; s >>= 1) {
    if (threadIdx.x < s) sm[threadIdx.x] += sm[threadIdx.x + s];
    __syncthreads();
  }
  if (threadIdx.x == 0) wsf[S1P_F + blockIdx.x] = sm[0];
}

// Pass 1b: per-block partial column sums of z (for ||sum z||^2 closed form).
__global__ void k_cols(const float* __restrict__ x, const float* __restrict__ y,
                       float* __restrict__ wsf) {
  const int d = threadIdx.x & 63, g = threadIdx.x >> 6;
  const int rbase = blockIdx.x * 256 + g * 64;
  float s = 0.f;
  for (int j = 0; j < 64; ++j) s += zrow(x, y, rbase + j)[d];
  __shared__ float sm[256];
  sm[threadIdx.x] = s;
  __syncthreads();
  if (threadIdx.x < 64)
    wsf[COLP_F + blockIdx.x * 64 + threadIdx.x] =
        sm[threadIdx.x] + sm[64 + threadIdx.x] + sm[128 + threadIdx.x] + sm[192 + threadIdx.x];
}

// Pass 2: bandwidth via closed form; store exp2 coefficient.
__global__ void k_bw(float* __restrict__ wsf) {
  const int t = threadIdx.x;  // 64 threads
  float cs = 0.f;
  for (int b = 0; b < 64; ++b) cs += wsf[COLP_F + b * 64 + t];
  __shared__ float smS[64], smQ[64];
  smS[t] = wsf[S1P_F + t];
  smQ[t] = cs * cs;
  __syncthreads();
  for (int s = 32; s > 0; s >>= 1) {
    if (t < s) { smS[t] += smS[t + s]; smQ[t] += smQ[t + s]; }
    __syncthreads();
  }
  if (t == 0) {
    float S1 = smS[0], SS = smQ[0];
    float N = (float)NTOT;
    float bw = (2.f * N * S1 - 2.f * SS) / (N * N - N);
    wsf[SCAL_F] = 1.f / (4.f * bw * 0.69314718055994531f);  // exp(-d2/(4bw)) = exp2(-d2*coef)
  }
}

// Main pass: one wave per 16x16 tile; hi/lo f16 split -> 3 WMMAs per K-step;
// fused multi-bandwidth RBF epilogue with quadrant-signed, symmetry-doubled weights.
__global__ void __launch_bounds__(256) k_mmd(const float* __restrict__ wsf,
                                             float* __restrict__ partial) {
  const int lane = threadIdx.x & 31;
  const int wv   = threadIdx.x >> 5;
  const int ti   = blockIdx.x;
  const int tj   = blockIdx.y * TJB + wv;
  float acc = 0.f;
  if (tj >= ti) {
    const int r  = lane & 15;
    const int hi = lane >> 4;
    const _Float16* zh = (const _Float16*)(wsf + ZH_F);
    const _Float16* zl = (const _Float16*)(wsf + ZL_F);
    const _Float16* Ah = zh + (size_t)(ti * 16 + r) * DIM;
    const _Float16* Al = zl + (size_t)(ti * 16 + r) * DIM;
    const _Float16* Bh = zh + (size_t)(tj * 16 + r) * DIM;
    const _Float16* Bl = zl + (size_t)(tj * 16 + r) * DIM;
    v8f c = {};
    #pragma unroll
    for (int kb = 0; kb < 2; ++kb) {
      // A 16x32 layout: lanes 0-15 hold K 0-15, lanes 16-31 hold K 16-31
      const int ka  = kb * 32 + hi * 16;
      // B 32x16 layout: lanes 0-15 hold K 0-15, lanes 16-31 hold K 16-31
      const int kbb = kb * 32 + hi * 16;
      v8h a0h = *(const v8h*)(Ah + ka);
      v8h a1h = *(const v8h*)(Ah + ka + 8);
      v8h a0l = *(const v8h*)(Al + ka);
      v8h a1l = *(const v8h*)(Al + ka + 8);
      v8h b0h = *(const v8h*)(Bh + kbb);
      v8h b1h = *(const v8h*)(Bh + kbb + 8);
      v8h b0l = *(const v8h*)(Bl + kbb);
      v8h b1l = *(const v8h*)(Bl + kbb + 8);
      v16h A_hi, A_lo, B_hi, B_lo;
      #pragma unroll
      for (int e = 0; e < 8; ++e) {
        A_hi[e] = a0h[e]; A_hi[8 + e] = a1h[e];
        A_lo[e] = a0l[e]; A_lo[8 + e] = a1l[e];
        B_hi[e] = b0h[e]; B_hi[8 + e] = b1h[e];
        B_lo[e] = b0l[e]; B_lo[8 + e] = b1l[e];
      }
      // gram ~= h*h + h*l + l*h  (drops l*l ~ 2^-22) -> near-f32 accuracy
      c = __builtin_amdgcn_wmma_f32_16x16x32_f16(false, A_hi, false, B_hi, (short)0, c, false, false);
      c = __builtin_amdgcn_wmma_f32_16x16x32_f16(false, A_hi, false, B_lo, (short)0, c, false, false);
      c = __builtin_amdgcn_wmma_f32_16x16x32_f16(false, A_lo, false, B_hi, (short)0, c, false, false);
    }
    const float coef = wsf[SCAL_F];
    const float sqj  = wsf[SQ_F + tj * 16 + r];
    const float4 s0  = *(const float4*)(wsf + SQ_F + ti * 16 + hi * 8);
    const float4 s1  = *(const float4*)(wsf + SQ_F + ti * 16 + hi * 8 + 4);
    const float sqi[8] = {s0.x, s0.y, s0.z, s0.w, s1.x, s1.y, s1.z, s1.w};
    float w = ((ti < NHALF / 16) == (tj < NHALF / 16)) ? 1.f : -1.f;  // quadrant sign
    if (tj > ti) w *= 2.f;                                            // symmetry
    float ks = 0.f;
    #pragma unroll
    for (int v = 0; v < 8; ++v) {
      float d2 = fmaxf(sqi[v] + sqj - 2.f * c[v], 0.f);
      float e  = FAST_EXP2(-d2 * coef);             // largest-bandwidth kernel, raw v_exp_f32
      float e2 = e * e, e4 = e2 * e2, e8 = e4 * e4, e16 = e8 * e8;
      ks += e + e2 + e4 + e8 + e16;                 // all 5 bandwidths
    }
    acc = w * ks;
  }
  #pragma unroll
  for (int o = 16; o > 0; o >>= 1) acc += __shfl_xor(acc, o, 32);
  __shared__ float sm[8];
  if (lane == 0) sm[wv] = acc;
  __syncthreads();
  if (threadIdx.x == 0) {
    float s = 0.f;
    #pragma unroll
    for (int q = 0; q < 8; ++q) s += sm[q];
    partial[blockIdx.x * gridDim.y + blockIdx.y] = s;
  }
}

// Deterministic two-stage reduction of block partials.
__global__ void k_red1(const float* __restrict__ partial, float* __restrict__ part2) {
  const int t = threadIdx.x;
  float v = partial[blockIdx.x * 512 + t] + partial[blockIdx.x * 512 + 256 + t];
  __shared__ float sm[256];
  sm[t] = v;
  __syncthreads();
  for (int s = 128; s > 0; s >>= 1) { if (t < s) sm[t] += sm[t + s]; __syncthreads(); }
  if (t == 0) part2[blockIdx.x] = sm[0];
}

__global__ void k_red2(const float* __restrict__ part2, float* __restrict__ out) {
  const int t = threadIdx.x;
  __shared__ float sm[256];
  sm[t] = part2[t];
  __syncthreads();
  for (int s = 128; s > 0; s >>= 1) { if (t < s) sm[t] += sm[t + s]; __syncthreads(); }
  if (t == 0) out[0] = sm[0] * (1.0f / ((float)NHALF * (float)NHALF));
}

extern "C" void kernel_launch(void* const* d_in, const int* in_sizes, int n_in,
                              void* d_out, int out_size, void* d_ws, size_t ws_size,
                              hipStream_t stream) {
  const float* x = (const float*)d_in[0];
  const float* y = (const float*)d_in[1];
  float* wsf = (float*)d_ws;
  float* out = (float*)d_out;

  k_rows<<<dim3(64), dim3(256), 0, stream>>>(x, y, wsf);
  k_cols<<<dim3(64), dim3(256), 0, stream>>>(x, y, wsf);
  k_bw  <<<dim3(1),  dim3(64),  0, stream>>>(wsf);
  k_mmd <<<dim3(TTILE, GRIDY), dim3(256), 0, stream>>>(wsf, wsf + PART_F);
  k_red1<<<dim3(256), dim3(256), 0, stream>>>(wsf + PART_F, wsf + PART2_F);
  k_red2<<<dim3(1),   dim3(256), 0, stream>>>(wsf + PART2_F, out);
}
